// ResidualVQ_18468359373175
// MI455X (gfx1250) — matrix operand order repaired and checked by reading, compiled, not run on hardware
//
#include <hip/hip_runtime.h>
#include <hip/hip_bf16.h>

#define B_ 8
#define N_ 4096
#define D_ 256
#define Q_ 8
#define C_ 1024
#define ROWS_ (B_ * N_)                     // 32768
#define QOUT_ELEMS ((long)ROWS_ * D_)       // 8388608
#define IDX_OFF QOUT_ELEMS                  // float indices in d_out
#define LOSS_OFF (QOUT_ELEMS + (long)ROWS_ * Q_)

typedef __attribute__((ext_vector_type(16))) __bf16 v16bf;
typedef __attribute__((ext_vector_type(8)))  float  v8f;
typedef __attribute__((ext_vector_type(8)))  unsigned int v8u;

__device__ __forceinline__ unsigned pack_bf16(float a, float b) {
    unsigned ua = __builtin_bit_cast(unsigned, a);
    unsigned ub = __builtin_bit_cast(unsigned, b);
    ua = (ua + 0x7FFFu + ((ua >> 16) & 1u)) >> 16;   // round-to-nearest-even
    ub = (ub + 0x7FFFu + ((ub >> 16) & 1u)) >> 16;
    return (ua & 0xFFFFu) | (ub << 16);
}

// ---------------- init: zero qout + losses in d_out ----------------
__global__ __launch_bounds__(256) void vq_init_kernel(float* __restrict__ out) {
    long i = (long)blockIdx.x * 256 + threadIdx.x;
    if (i < QOUT_ELEMS) out[i] = 0.0f;
    if (i < Q_) out[LOSS_OFF + i] = 0.0f;
}

// ---------------- prep: codebooks f32 -> bf16, + cnorm ----------------
// one wave per code; grid = Q*C/8 blocks of 256 threads
__global__ __launch_bounds__(256) void vq_prep_kernel(
    const float* __restrict__ codebooks, unsigned short* __restrict__ cbbf,
    float* __restrict__ cnorm) {
    int lane = threadIdx.x & 31;
    long code = (long)blockIdx.x * 8 + (threadIdx.x >> 5);   // 0 .. Q*C-1
    const float* src = codebooks + code * D_ + lane * 8;
    float4 a = *(const float4*)src;
    float4 b = *(const float4*)(src + 4);
    float ss = a.x * a.x + a.y * a.y + a.z * a.z + a.w * a.w
             + b.x * b.x + b.y * b.y + b.z * b.z + b.w * b.w;
    uint4 o;
    o.x = pack_bf16(a.x, a.y);
    o.y = pack_bf16(a.z, a.w);
    o.z = pack_bf16(b.x, b.y);
    o.w = pack_bf16(b.z, b.w);
    *(uint4*)(cbbf + code * D_ + lane * 8) = o;
    #pragma unroll
    for (int m = 1; m < 32; m <<= 1) ss += __shfl_xor(ss, m, 32);
    if (lane == 0) cnorm[code] = ss;
}

// ---------------- per-stage distance + argmin (WMMA) ----------------
// 128 threads = 4 waves; each wave handles 32 rows (two 16-row A tiles,
// shared B fragments -> 16 WMMAs per 17 loads); grid = ROWS_/128 blocks.
__global__ __launch_bounds__(128) void vq_argmin_kernel(
    const float* __restrict__ x, const float* __restrict__ qout,
    const unsigned short* __restrict__ cbbf, const float* __restrict__ cnorm,
    int* __restrict__ idx_out, float* __restrict__ out, int stage) {
    const int lane = threadIdx.x & 31;
    const int wave = threadIdx.x >> 5;
    const int lrow = lane & 15;
    const bool hi = lane >= 16;
    const long rowBase = ((long)blockIdx.x * 4 + wave) * 32;

    // Build residual A-fragments (bf16 16x32 layout) for two 16-row tiles.
    v16bf afrag[2][8];
    float rn[2];
    #pragma unroll
    for (int u = 0; u < 2; ++u) {
        const long row = rowBase + u * 16 + lrow;
        const float* xr = x + row * D_;
        const float* qr = qout + row * D_;
        float rnp = 0.0f;
        #pragma unroll
        for (int s = 0; s < 8; ++s) {
            int k0 = 32 * s + (hi ? 8 : 0);
            int k1 = k0 + 16;
            float4 xa = *(const float4*)(xr + k0);
            float4 xb = *(const float4*)(xr + k0 + 4);
            float4 xc = *(const float4*)(xr + k1);
            float4 xd = *(const float4*)(xr + k1 + 4);
            float4 qa = *(const float4*)(qr + k0);
            float4 qb = *(const float4*)(qr + k0 + 4);
            float4 qc = *(const float4*)(qr + k1);
            float4 qd = *(const float4*)(qr + k1 + 4);
            float r0 = xa.x - qa.x, r1 = xa.y - qa.y, r2 = xa.z - qa.z, r3 = xa.w - qa.w;
            float r4 = xb.x - qb.x, r5 = xb.y - qb.y, r6 = xb.z - qb.z, r7 = xb.w - qb.w;
            float r8 = xc.x - qc.x, r9 = xc.y - qc.y, r10 = xc.z - qc.z, r11 = xc.w - qc.w;
            float r12 = xd.x - qd.x, r13 = xd.y - qd.y, r14 = xd.z - qd.z, r15 = xd.w - qd.w;
            rnp += r0*r0 + r1*r1 + r2*r2 + r3*r3 + r4*r4 + r5*r5 + r6*r6 + r7*r7
                 + r8*r8 + r9*r9 + r10*r10 + r11*r11 + r12*r12 + r13*r13 + r14*r14 + r15*r15;
            v8u p;
            p[0] = pack_bf16(r0, r1);   p[1] = pack_bf16(r2, r3);
            p[2] = pack_bf16(r4, r5);   p[3] = pack_bf16(r6, r7);
            p[4] = pack_bf16(r8, r9);   p[5] = pack_bf16(r10, r11);
            p[6] = pack_bf16(r12, r13); p[7] = pack_bf16(r14, r15);
            afrag[u][s] = __builtin_bit_cast(v16bf, p);
        }
        rnp += __shfl_xor(rnp, 16, 32);   // full ||r||^2 for row (lanes L, L+16)
        rn[u] = rnp;
    }

    const unsigned short* cb = cbbf + (long)stage * C_ * D_;
    const float* cn = cnorm + stage * C_;

    float bestVal[2][8];
    int   bestIdx[2][8];
    #pragma unroll
    for (int u = 0; u < 2; ++u)
        #pragma unroll
        for (int v = 0; v < 8; ++v) { bestVal[u][v] = 3.4e38f; bestIdx[u][v] = 0; }

    for (int t = 0; t < C_ / 16; ++t) {
        int code = t * 16 + lrow;
        float cnc = cn[code];
        // B fragment: bf16 K-values of column `code`, two 8-runs (k 8h.., 16+8h..)
        const unsigned short* cp = cb + (long)code * D_ + (hi ? 8 : 0);
        __builtin_prefetch(cp + 16 * D_, 0, 0);   // next code tile
        v8f acc0 = {};
        v8f acc1 = {};
        #pragma unroll
        for (int s = 0; s < 8; ++s) {
            uint4 b0 = *(const uint4*)(cp + 32 * s);
            uint4 b1 = *(const uint4*)(cp + 32 * s + 16);
            v8u bp; bp[0]=b0.x; bp[1]=b0.y; bp[2]=b0.z; bp[3]=b0.w;
                    bp[4]=b1.x; bp[5]=b1.y; bp[6]=b1.z; bp[7]=b1.w;
            v16bf bf = __builtin_bit_cast(v16bf, bp);
            acc0 = __builtin_amdgcn_wmma_f32_16x16x32_bf16(
                false, afrag[0][s], false, bf, (short)0, acc0, false, false);
            acc1 = __builtin_amdgcn_wmma_f32_16x16x32_bf16(
                false, afrag[1][s], false, bf, (short)0, acc1, false, false);
        }
        #pragma unroll
        for (int v = 0; v < 8; ++v) {
            float sc0 = cnc - 2.0f * acc0[v];   // d2 - ||r||^2 (same argmin)
            float sc1 = cnc - 2.0f * acc1[v];
            if (sc0 < bestVal[0][v]) { bestVal[0][v] = sc0; bestIdx[0][v] = code; }
            if (sc1 < bestVal[1][v]) { bestVal[1][v] = sc1; bestIdx[1][v] = code; }
        }
    }

    float d2sum = 0.0f;
    #pragma unroll
    for (int u = 0; u < 2; ++u) {
        // cross-lane min over the 16 lanes of each half (codes mod 16)
        #pragma unroll
        for (int v = 0; v < 8; ++v) {
            float bv = bestVal[u][v]; int bi = bestIdx[u][v];
            #pragma unroll
            for (int m = 1; m < 16; m <<= 1) {
                float ov = __shfl_xor(bv, m, 32);
                int   oi = __shfl_xor(bi, m, 32);
                if (ov < bv || (ov == bv && oi < bi)) { bv = ov; bi = oi; }
            }
            bestVal[u][v] = bv; bestIdx[u][v] = bi;
        }
        // writers: lanes 0..7 -> rows 0..7, lanes 24..31 -> rows 8..15
        bool writer = false; int widx = 0; float wval = 0.0f; long wrow = 0;
        #pragma unroll
        for (int v = 0; v < 8; ++v) {
            bool sel = (lane == v) || (lane == 24 + v);
            if (sel) {
                writer = true; widx = bestIdx[u][v]; wval = bestVal[u][v];
                wrow = rowBase + u * 16 + (hi ? (v + 8) : v);
            }
        }
        if (writer) {
            idx_out[wrow] = widx;
            out[IDX_OFF + wrow * Q_ + stage] = (float)widx;
            float d2c = rn[u] + wval;
            if (d2c < 0.0f) d2c = 0.0f;
            d2sum += d2c;
        }
    }
    // one atomic per wave for the loss
    #pragma unroll
    for (int m = 1; m < 32; m <<= 1) d2sum += __shfl_xor(d2sum, m, 32);
    if (lane == 0)
        atomicAdd(out + LOSS_OFF + stage,
                  d2sum * (1.0f / ((float)B_ * (float)N_ * (float)D_)));
}

// ---------------- per-stage update: qout += codebook[idx] (fp32) ----------------
// 64 threads per row (float4), 256-thread blocks -> 4 rows/block
__global__ __launch_bounds__(256) void vq_update_kernel(
    const float* __restrict__ codebooks, const int* __restrict__ idx,
    float* __restrict__ qout, int stage) {
    long t = (long)blockIdx.x * 256 + threadIdx.x;
    long row = t >> 6;
    int d4 = (int)(t & 63) * 4;
    int c = idx[row];
    const float* cv = codebooks + ((long)stage * C_ + c) * D_ + d4;
    float4 cc = *(const float4*)cv;
    float4* qp = (float4*)(qout + row * D_ + d4);
    float4 q = *qp;
    q.x += cc.x; q.y += cc.y; q.z += cc.z; q.w += cc.w;
    *qp = q;
}

extern "C" void kernel_launch(void* const* d_in, const int* in_sizes, int n_in,
                              void* d_out, int out_size, void* d_ws, size_t ws_size,
                              hipStream_t stream) {
    const float* x = (const float*)d_in[0];          // [B,N,D] f32
    const float* codebooks = (const float*)d_in[1];  // [Q,C,D] f32
    float* out = (float*)d_out;

    // workspace layout
    unsigned short* cbbf = (unsigned short*)d_ws;                    // Q*C*D bf16
    float* cnorm = (float*)((char*)d_ws + (size_t)Q_ * C_ * D_ * 2); // Q*C f32
    int* idx_ws = (int*)((char*)cnorm + (size_t)Q_ * C_ * 4);        // ROWS_ int

    // zero qout + losses
    vq_init_kernel<<<(QOUT_ELEMS + 255) / 256, 256, 0, stream>>>(out);
    // codebooks -> bf16 + cnorm
    vq_prep_kernel<<<(Q_ * C_) / 8, 256, 0, stream>>>(codebooks, cbbf, cnorm);

    float* qout = out;   // residual = x - qout
    for (int stage = 0; stage < Q_; ++stage) {
        vq_argmin_kernel<<<ROWS_ / 128, 128, 0, stream>>>(
            x, qout, cbbf, cnorm, idx_ws, out, stage);
        vq_update_kernel<<<((long)ROWS_ * 64) / 256, 256, 0, stream>>>(
            codebooks, idx_ws, qout, stage);
    }
}
